// MegaBlockFeedForward_44890998177888
// MI455X (gfx1250) — compile-verified, matrix operand order checked by reading, not run on hardware
//
#include <hip/hip_runtime.h>
#include <hip/hip_bf16.h>

// ---------------------------------------------------------------------------
// MoE SwiGLU FFN for MI455X (gfx1250, wave32, WMMA).
// fp32 -> bf16 hi/lo truncation split ("bf16x3") on V_WMMA_F32_16X16X32_BF16,
// fp32 accumulation. Conversion happens once per LDS tile (not per fragment),
// so the inner loop is ds_load_b128 + v_wmma only.
// Stage 1: g = silu(x@w1^T)*(x@w2^T), written to d_ws as bf16 hi/lo planes.
// Stage 2: out = g @ w3; the A operand (bf16 planes, pure copy) streams into
//          double-buffered LDS via GLOBAL_LOAD_ASYNC_TO_LDS_B128 (ASYNCcnt).
// ---------------------------------------------------------------------------

typedef __attribute__((ext_vector_type(16))) __bf16 v16bf;
typedef __attribute__((ext_vector_type(8)))  __bf16 v8bf;
typedef __attribute__((ext_vector_type(8)))  float  v8f;

union V16 { v16bf v; v8bf h[2]; };

__device__ __forceinline__ unsigned int fbits(float f) {
  return __builtin_bit_cast(unsigned int, f);
}
__device__ __forceinline__ float bcastf(unsigned int u) {
  return __builtin_bit_cast(float, u);
}

// Pack the top halves of two fp32 words into one dword: {u1.hi16, u0.hi16}.
__device__ __forceinline__ unsigned int pack_hi16(unsigned int u0, unsigned int u1) {
#if __has_builtin(__builtin_amdgcn_perm)
  return __builtin_amdgcn_perm(u1, u0, 0x07060302u);  // v_perm_b32
#else
  return (u0 >> 16) | (u1 & 0xffff0000u);
#endif
}

// Truncation split of 4 fp32 into packed bf16 hi and lo planes (2 dwords each).
struct Pack4 { unsigned int hi0, hi1, lo0, lo1; };
__device__ __forceinline__ Pack4 split4(float4 f) {
  unsigned int u0 = fbits(f.x), u1 = fbits(f.y), u2 = fbits(f.z), u3 = fbits(f.w);
  float l0 = f.x - bcastf(u0 & 0xffff0000u);
  float l1 = f.y - bcastf(u1 & 0xffff0000u);
  float l2 = f.z - bcastf(u2 & 0xffff0000u);
  float l3 = f.w - bcastf(u3 & 0xffff0000u);
  Pack4 p;
  p.hi0 = pack_hi16(u0, u1);
  p.hi1 = pack_hi16(u2, u3);
  p.lo0 = pack_hi16(fbits(l0), fbits(l1));
  p.lo1 = pack_hi16(fbits(l2), fbits(l3));
  return p;
}

// Low 32 bits of a generic pointer to __shared__ = LDS byte address.
__device__ __forceinline__ unsigned int lds_addr_of(const void* p) {
  return (unsigned int)(unsigned long long)p;
}

// Async global->LDS copy of 16 bytes (per lane), tracked with ASYNCcnt.
__device__ __forceinline__ void async_copy_b128(unsigned int ldsDst, const void* gsrc) {
  asm volatile("global_load_async_to_lds_b128 %0, %1, off"
               :: "v"(ldsDst), "v"(gsrc) : "memory");
}
__device__ __forceinline__ void wait_async_le4() {
#if __has_builtin(__builtin_amdgcn_s_wait_asynccnt)
  __builtin_amdgcn_s_wait_asynccnt(4);
#else
  asm volatile("s_wait_asynccnt 0x4" ::: "memory");
#endif
}
__device__ __forceinline__ void wait_async_0() {
#if __has_builtin(__builtin_amdgcn_s_wait_asynccnt)
  __builtin_amdgcn_s_wait_asynccnt(0);
#else
  asm volatile("s_wait_asynccnt 0x0" ::: "memory");
#endif
}

// Load one 16x32 bf16 fragment from an LDS tile (stride S shorts).
// CDNA5 16-bit A/B layout: lanes 0-15 entity=entBase+lane, K {0..7,16..23};
// lanes 16-31 entity=entBase+lane-16, K {8..15,24..31}.
// v16bf slot s <-> K = ks + s + (s>=8 ? 8 : 0)  => two contiguous 8-elem runs.
__device__ __forceinline__ v16bf frag_bf16(const unsigned short* tile,
                                           int entBase, int S) {
  const int lane = (int)(threadIdx.x & 31u);
  const int ks   = (lane & 16) ? 8 : 0;
  const unsigned short* p = tile + (entBase + (lane & 15)) * S;
  V16 u;
  u.h[0] = *reinterpret_cast<const v8bf*>(p + ks);        // K ks..ks+7
  u.h[1] = *reinterpret_cast<const v8bf*>(p + ks + 16);   // K ks+16..ks+23
  return u.v;
}

// One logical fp32 16x16x32 MMA via three bf16 WMMAs (drop lo*lo term).
__device__ __forceinline__ v8f mma_bf16x3(v8f acc,
                                          const v16bf ah, const v16bf al,
                                          const v16bf bh, const v16bf bl) {
  acc = __builtin_amdgcn_wmma_f32_16x16x32_bf16(false, ah, false, bh, (short)0, acc, false, false);
  acc = __builtin_amdgcn_wmma_f32_16x16x32_bf16(false, ah, false, bl, (short)0, acc, false, false);
  acc = __builtin_amdgcn_wmma_f32_16x16x32_bf16(false, al, false, bh, (short)0, acc, false, false);
  return acc;
}

#define STRIDE 40  // bf16 tile row stride (shorts): 80B rows, 16B aligned, staggered banks

// ---------------------------------------------------------------------------
// Stage 1: h1 = xe @ w1e^T, h2 = xe @ w2e^T, g = silu(h1)*h2.
// Block tile 128 tokens x 64 hidden, 8 waves (4Mx2N), wave tile 32x32 per matrix.
// ---------------------------------------------------------------------------
__global__ __launch_bounds__(256) void moe_stage1(const float* __restrict__ x,
                                                  const float* __restrict__ w1,
                                                  const float* __restrict__ w2,
                                                  unsigned short* __restrict__ ghi,
                                                  unsigned short* __restrict__ glo) {
  __shared__ unsigned short sAhi[128 * STRIDE], sAlo[128 * STRIDE];   // 10 KB each
  __shared__ unsigned short sB1hi[64 * STRIDE], sB1lo[64 * STRIDE];   //  5 KB each
  __shared__ unsigned short sB2hi[64 * STRIDE], sB2lo[64 * STRIDE];

  const int e    = blockIdx.z;
  const int mblk = blockIdx.y;
  const int nblk = blockIdx.x;
  const int tid  = (int)threadIdx.x;
  const int lane = tid & 31;
  const int wid  = tid >> 5;
  const int waveM = wid >> 1;      // 0..3
  const int waveN = wid & 1;       // 0..1

  const long long tok0  = (long long)e * 2048 + (long long)mblk * 128;
  const int       h0    = nblk * 64;
  const long long wrow0 = (long long)e * 1024 + h0;

  float4 ra[4], rb1[2], rb2[2];

  auto loadTiles = [&](int kk) {
#pragma unroll
    for (int i = 0; i < 4; ++i) {
      int c = tid + i * 256;
      int r = c >> 3, c4 = (c & 7) << 2;
      ra[i] = *reinterpret_cast<const float4*>(x + (tok0 + r) * 2048 + kk + c4);
    }
#pragma unroll
    for (int i = 0; i < 2; ++i) {
      int c = tid + i * 256;
      int r = c >> 3, c4 = (c & 7) << 2;
      rb1[i] = *reinterpret_cast<const float4*>(w1 + (wrow0 + r) * 2048 + kk + c4);
      rb2[i] = *reinterpret_cast<const float4*>(w2 + (wrow0 + r) * 2048 + kk + c4);
    }
  };
  auto storeTiles = [&]() {
#pragma unroll
    for (int i = 0; i < 4; ++i) {
      int c = tid + i * 256;
      int r = c >> 3, c4 = (c & 7) << 2;
      Pack4 p = split4(ra[i]);
      *reinterpret_cast<uint2*>(sAhi + r * STRIDE + c4) = make_uint2(p.hi0, p.hi1);
      *reinterpret_cast<uint2*>(sAlo + r * STRIDE + c4) = make_uint2(p.lo0, p.lo1);
    }
#pragma unroll
    for (int i = 0; i < 2; ++i) {
      int c = tid + i * 256;
      int r = c >> 3, c4 = (c & 7) << 2;
      Pack4 p1 = split4(rb1[i]);
      *reinterpret_cast<uint2*>(sB1hi + r * STRIDE + c4) = make_uint2(p1.hi0, p1.hi1);
      *reinterpret_cast<uint2*>(sB1lo + r * STRIDE + c4) = make_uint2(p1.lo0, p1.lo1);
      Pack4 p2 = split4(rb2[i]);
      *reinterpret_cast<uint2*>(sB2hi + r * STRIDE + c4) = make_uint2(p2.hi0, p2.hi1);
      *reinterpret_cast<uint2*>(sB2lo + r * STRIDE + c4) = make_uint2(p2.lo0, p2.lo1);
    }
  };

  v8f acc1[2][2] = {};
  v8f acc2[2][2] = {};

  loadTiles(0);
  for (int kk = 0; kk < 2048; kk += 32) {
    __syncthreads();
    storeTiles();
    __syncthreads();
    if (kk + 32 < 2048) loadTiles(kk + 32);  // prefetch next K-step into regs

    v16bf aHi[2], aLo[2];
#pragma unroll
    for (int mt = 0; mt < 2; ++mt) {
      aHi[mt] = frag_bf16(sAhi, waveM * 32 + mt * 16, STRIDE);
      aLo[mt] = frag_bf16(sAlo, waveM * 32 + mt * 16, STRIDE);
    }
    {  // w1 path
      v16bf bHi[2], bLo[2];
#pragma unroll
      for (int nt = 0; nt < 2; ++nt) {
        bHi[nt] = frag_bf16(sB1hi, waveN * 32 + nt * 16, STRIDE);
        bLo[nt] = frag_bf16(sB1lo, waveN * 32 + nt * 16, STRIDE);
      }
#pragma unroll
      for (int mt = 0; mt < 2; ++mt)
#pragma unroll
        for (int nt = 0; nt < 2; ++nt)
          acc1[mt][nt] = mma_bf16x3(acc1[mt][nt], aHi[mt], aLo[mt], bHi[nt], bLo[nt]);
    }
    {  // w2 path (reuses B fragment registers)
      v16bf bHi[2], bLo[2];
#pragma unroll
      for (int nt = 0; nt < 2; ++nt) {
        bHi[nt] = frag_bf16(sB2hi, waveN * 32 + nt * 16, STRIDE);
        bLo[nt] = frag_bf16(sB2lo, waveN * 32 + nt * 16, STRIDE);
      }
#pragma unroll
      for (int mt = 0; mt < 2; ++mt)
#pragma unroll
        for (int nt = 0; nt < 2; ++nt)
          acc2[mt][nt] = mma_bf16x3(acc2[mt][nt], aHi[mt], aLo[mt], bHi[nt], bLo[nt]);
    }
  }

  // Epilogue: g = silu(h1)*h2, written as bf16 hi/lo planes for stage 2.
  // C/D layout: lanes<16 -> M=r, lanes>=16 -> M=8+r; N = lane&15.
  const int laneN  = lane & 15;
  const int rowSel = (lane >> 4) << 3;
#pragma unroll
  for (int mt = 0; mt < 2; ++mt)
#pragma unroll
    for (int nt = 0; nt < 2; ++nt) {
      const long long col = h0 + waveN * 32 + nt * 16 + laneN;
#pragma unroll
      for (int r = 0; r < 8; ++r) {
        float h1v = acc1[mt][nt][r];
        float h2v = acc2[mt][nt][r];
        float s   = h1v / (1.0f + __expf(-h1v));  // silu
        float gv  = s * h2v;
        unsigned int u = fbits(gv);
        float lof = gv - bcastf(u & 0xffff0000u);
        long long row = tok0 + waveM * 32 + mt * 16 + rowSel + r;
        ghi[row * 1024 + col] = (unsigned short)(u >> 16);
        glo[row * 1024 + col] = (unsigned short)(fbits(lof) >> 16);
      }
    }
}

// ---------------------------------------------------------------------------
// Stage 2: out = g @ w3e  (per expert [2048x1024]@[1024x2048]).
// A operand (bf16 hi/lo planes) streams straight into double-buffered LDS via
// async copies (no VGPR transit). Async loads complete in order, so waiting
// ASYNCcnt<=4 retires exactly the current buffer's 4 copies per thread while
// the next buffer's copies remain in flight.  w3 is split+transposed at store.
// ---------------------------------------------------------------------------
__global__ __launch_bounds__(256) void moe_stage2(const unsigned short* __restrict__ ghi,
                                                  const unsigned short* __restrict__ glo,
                                                  const float* __restrict__ w3,
                                                  float* __restrict__ out) {
  __shared__ unsigned short sAhi[2][128 * STRIDE], sAlo[2][128 * STRIDE];  // 2 x 20 KB
  __shared__ unsigned short sBhi[64 * STRIDE],     sBlo[64 * STRIDE];      // 10 KB

  const int e    = blockIdx.z;
  const int mblk = blockIdx.y;
  const int nblk = blockIdx.x;
  const int tid  = (int)threadIdx.x;
  const int lane = tid & 31;
  const int wid  = tid >> 5;
  const int waveM = wid >> 1;
  const int waveN = wid & 1;

  const long long tok0 = (long long)e * 2048 + (long long)mblk * 128;
  const int d0 = nblk * 64;

  float4 rb[2];

  auto issueAsyncA = [&](int kk, int buf) {
#pragma unroll
    for (int i = 0; i < 2; ++i) {           // A: 128x32 bf16, 16B chunks
      int c = tid + i * 256;
      int r = c >> 2, e0 = (c & 3) << 3;    // 4 chunks per row, 8 shorts each
      async_copy_b128(lds_addr_of(&sAhi[buf][r * STRIDE + e0]),
                      ghi + (tok0 + r) * 1024 + kk + e0);
      async_copy_b128(lds_addr_of(&sAlo[buf][r * STRIDE + e0]),
                      glo + (tok0 + r) * 1024 + kk + e0);
    }
  };
  auto loadB = [&](int kk) {
#pragma unroll
    for (int i = 0; i < 2; ++i) {           // B: w3 rows=h(K), cols=d(N)
      int c = tid + i * 256;
      int r = c >> 4, c4 = (c & 15) << 2;
      rb[i] = *reinterpret_cast<const float4*>(
          w3 + ((long long)e * 1024 + kk + r) * 2048 + d0 + c4);
    }
  };
  auto storeB = [&]() {
#pragma unroll
    for (int i = 0; i < 2; ++i) {           // split + transpose: [d][h]
      int c = tid + i * 256;
      int r = c >> 4, c4 = (c & 15) << 2;
      Pack4 p = split4(rb[i]);
      sBhi[(c4 + 0) * STRIDE + r] = (unsigned short)(p.hi0);
      sBhi[(c4 + 1) * STRIDE + r] = (unsigned short)(p.hi0 >> 16);
      sBhi[(c4 + 2) * STRIDE + r] = (unsigned short)(p.hi1);
      sBhi[(c4 + 3) * STRIDE + r] = (unsigned short)(p.hi1 >> 16);
      sBlo[(c4 + 0) * STRIDE + r] = (unsigned short)(p.lo0);
      sBlo[(c4 + 1) * STRIDE + r] = (unsigned short)(p.lo0 >> 16);
      sBlo[(c4 + 2) * STRIDE + r] = (unsigned short)(p.lo1);
      sBlo[(c4 + 3) * STRIDE + r] = (unsigned short)(p.lo1 >> 16);
    }
  };

  v8f acc[2][2] = {};

  issueAsyncA(0, 0);   // prime A ping-pong
  loadB(0);            // prime B register prefetch
  int buf = 0;
  for (int kk = 0; kk < 1024; kk += 32) {
    const bool hasNext = (kk + 32 < 1024);
    __syncthreads();                        // prior compute done: sB + buf^1 free
    storeB();
    if (hasNext) {
      issueAsyncA(kk + 32, buf ^ 1);        // next A buffer in flight
      wait_async_le4();                     // retire current buffer (in-order)
    } else {
      wait_async_0();
    }
    __syncthreads();                        // publish sB stores + A copies
    if (hasNext) loadB(kk + 32);            // B regs refill during compute

    v16bf aHi[2], aLo[2], bHi[2], bLo[2];
#pragma unroll
    for (int mt = 0; mt < 2; ++mt) {
      aHi[mt] = frag_bf16(sAhi[buf], waveM * 32 + mt * 16, STRIDE);
      aLo[mt] = frag_bf16(sAlo[buf], waveM * 32 + mt * 16, STRIDE);
    }
#pragma unroll
    for (int nt = 0; nt < 2; ++nt) {
      bHi[nt] = frag_bf16(sBhi, waveN * 32 + nt * 16, STRIDE);
      bLo[nt] = frag_bf16(sBlo, waveN * 32 + nt * 16, STRIDE);
    }
#pragma unroll
    for (int mt = 0; mt < 2; ++mt)
#pragma unroll
      for (int nt = 0; nt < 2; ++nt)
        acc[mt][nt] = mma_bf16x3(acc[mt][nt], aHi[mt], aLo[mt], bHi[nt], bLo[nt]);

    buf ^= 1;
  }

  const int laneN  = lane & 15;
  const int rowSel = (lane >> 4) << 3;
#pragma unroll
  for (int mt = 0; mt < 2; ++mt)
#pragma unroll
    for (int nt = 0; nt < 2; ++nt) {
      const long long col = d0 + waveN * 32 + nt * 16 + laneN;
#pragma unroll
      for (int r = 0; r < 8; ++r) {
        long long row = tok0 + waveM * 32 + mt * 16 + rowSel + r;
        out[row * 2048 + col] = acc[mt][nt][r];
      }
    }
}

// ---------------------------------------------------------------------------
extern "C" void kernel_launch(void* const* d_in, const int* in_sizes, int n_in,
                              void* d_out, int out_size, void* d_ws, size_t ws_size,
                              hipStream_t stream) {
  const float* x  = (const float*)d_in[0];
  const float* w1 = (const float*)d_in[1];
  const float* w2 = (const float*)d_in[2];
  const float* w3 = (const float*)d_in[3];
  // d_in[4] = num_experts (8), baked into the grid (z dimension).
  float* out = (float*)d_out;

  // Workspace: g as bf16 hi/lo planes, 16384x1024 each (64 MB total).
  unsigned short* ghi = (unsigned short*)d_ws;
  unsigned short* glo = ghi + (size_t)16384 * 1024;

  dim3 blk(256);
  moe_stage1<<<dim3(16, 16, 8), blk, 0, stream>>>(x, w1, w2, ghi, glo);
  moe_stage2<<<dim3(32, 16, 8), blk, 0, stream>>>(ghi, glo, w3, out);
}
